// RotaryEmbeddingTransformerLayer_66365834658503
// MI455X (gfx1250) — matrix operand order repaired and checked by reading, not compiled, not run on hardware
//
#include <hip/hip_runtime.h>
#include <hip/hip_bf16.h>

// ---------------------------------------------------------------------------
// Types for CDNA5 WMMA (wave32, 16x16x32 bf16, f32 accumulate)
// ---------------------------------------------------------------------------
typedef __attribute__((ext_vector_type(16))) __bf16   v16bf;
typedef __attribute__((ext_vector_type(8)))  float    v8f;
typedef __attribute__((ext_vector_type(4)))  float    v4f;
typedef __attribute__((ext_vector_type(4)))  unsigned int v4u;

union Frag16 {
    v16bf v;
    unsigned int u[8];
    v4u q[2];
};

__device__ __forceinline__ v8f wmma_bf16(v16bf a, v16bf b, v8f c) {
    // (neg_a, A, neg_b, B, c_mod, C, reuse_a, reuse_b)
    return __builtin_amdgcn_wmma_f32_16x16x32_bf16(false, a, false, b,
                                                   (short)0, c, false, false);
}

__device__ __forceinline__ unsigned short f2bf(float f) {
    unsigned int u = __float_as_uint(f);
    u += 0x7FFFu + ((u >> 16) & 1u);   // round-to-nearest-even
    return (unsigned short)(u >> 16);
}
__device__ __forceinline__ float bf2f(unsigned short h) {
    return __uint_as_float(((unsigned int)h) << 16);
}

// ---------------------------------------------------------------------------
// CDNA5 async global->LDS staging (ASYNCcnt DMA), guarded so the file still
// compiles on the host pass / toolchains without the builtin.
// Builtin signature (from hipcc diagnostic): param0 = int4-vector ptr in
// addrspace(1) ("__device__"), param1 = LDS side, 2 imm ints (offset, cpol).
// ---------------------------------------------------------------------------
#if defined(__AMDGCN__) && __has_builtin(__builtin_amdgcn_global_load_async_to_lds_b128)
#define HAVE_ASYNC_LDS 1
#else
#define HAVE_ASYNC_LDS 0
#endif

#if HAVE_ASYNC_LDS
typedef int v4i_b128 __attribute__((vector_size(4 * sizeof(int))));
typedef __attribute__((address_space(1))) v4i_b128 as1_v4i;
typedef __attribute__((address_space(3))) v4i_b128 as3_v4i;
__device__ __forceinline__ void async_b128(const void* g, void* l) {
    __builtin_amdgcn_global_load_async_to_lds_b128(
        (as1_v4i*)const_cast<void*>(g), (as3_v4i*)l, 0, 0);
}
__device__ __forceinline__ void wait_async0() {
#if __has_builtin(__builtin_amdgcn_s_wait_asynccnt)
    __builtin_amdgcn_s_wait_asynccnt(0);
#else
    asm volatile("s_wait_asynccnt 0" ::: "memory");
#endif
}
#endif

#define D_MODEL 512
#define N_HEADS 8
#define HEAD_DIM 64
#define D_FF 2048
#define SEQ 4096
#define BATCH 2
#define MROWS (BATCH * SEQ)   // 8192

// ---------------------------------------------------------------------------
// Weight fp32 [N][K] -> bf16 transposed [K][N]
// ---------------------------------------------------------------------------
__global__ __launch_bounds__(256) void cvtT_kernel(
    const float* __restrict__ w, unsigned short* __restrict__ wt, int N, int K)
{
    int idx = blockIdx.x * 256 + threadIdx.x;
    if (idx >= N * K) return;
    int n = idx / K, k = idx - n * K;
    wt[(size_t)k * N + n] = f2bf(w[idx]);
}

// ---------------------------------------------------------------------------
// LayerNorm: one wave per 512-wide row, fp32 in -> bf16 out
// ---------------------------------------------------------------------------
__global__ __launch_bounds__(256) void ln_kernel(
    const float* __restrict__ x, const float* __restrict__ w,
    const float* __restrict__ b, unsigned short* __restrict__ y, int rows)
{
    int wid  = (blockIdx.x * blockDim.x + threadIdx.x) >> 5;
    int lane = threadIdx.x & 31;
    if (wid >= rows) return;

    const v4f* xr = (const v4f*)(x + (size_t)wid * D_MODEL);
    v4f xs[4];
    float s = 0.f, s2 = 0.f;
#pragma unroll
    for (int i = 0; i < 4; i++) {
        xs[i] = xr[lane * 4 + i];
#pragma unroll
        for (int c = 0; c < 4; c++) { s += xs[i][c]; s2 += xs[i][c] * xs[i][c]; }
    }
#pragma unroll
    for (int m = 16; m >= 1; m >>= 1) {
        s  += __shfl_xor(s,  m);
        s2 += __shfl_xor(s2, m);
    }
    float mu  = s * (1.f / D_MODEL);
    float var = s2 * (1.f / D_MODEL) - mu * mu;
    float inv = rsqrtf(var + 1e-5f);

    unsigned int pk[8];
#pragma unroll
    for (int i = 0; i < 4; i++) {
        v4f wv = ((const v4f*)w)[lane * 4 + i];
        v4f bv = ((const v4f*)b)[lane * 4 + i];
        unsigned short h[4];
#pragma unroll
        for (int c = 0; c < 4; c++)
            h[c] = f2bf((xs[i][c] - mu) * inv * wv[c] + bv[c]);
        pk[i * 2]     = (unsigned int)h[0] | ((unsigned int)h[1] << 16);
        pk[i * 2 + 1] = (unsigned int)h[2] | ((unsigned int)h[3] << 16);
    }
    v4u* out = (v4u*)(y + (size_t)wid * D_MODEL + lane * 16);
    v4u o0 = {pk[0], pk[1], pk[2], pk[3]};
    v4u o1 = {pk[4], pk[5], pk[6], pk[7]};
    out[0] = o0; out[1] = o1;
}

// ---------------------------------------------------------------------------
// WMMA GEMM: C[M,N] = A[M,K](bf16) * Wt[K,N](bf16)  (+bias, +resid, relu)
// 256 threads = 8 waves; block tile 128x128; wave tile 64x32.
// Double-buffered ASYNCcnt global->LDS staging on CDNA5.
// ---------------------------------------------------------------------------
template<bool RELU, bool RESID, bool OUT_BF16>
__global__ __launch_bounds__(256) void gemm_bf16_kernel(
    const unsigned short* __restrict__ A,
    const unsigned short* __restrict__ Bt,
    const float* __restrict__ bias,
    const float* __restrict__ resid,
    void* __restrict__ Cout, int M, int N, int K)
{
    __shared__ __align__(16) unsigned short lA[2][128 * 32];
    __shared__ __align__(16) unsigned short lB[2][32 * 128];

    const int tid  = threadIdx.x;
    const int lane = tid & 31;
    const int wave = tid >> 5;
    const int lh   = lane >> 4;      // half-wave select
    const int l16  = lane & 15;
    const int bM = blockIdx.y * 128;
    const int bN = blockIdx.x * 128;
    const int wM = (wave >> 2) * 64; // 2 waves along M
    const int wN = (wave & 3) * 32;  // 4 waves along N

    v8f acc[4][2];
#pragma unroll
    for (int i = 0; i < 4; i++)
#pragma unroll
        for (int j = 0; j < 2; j++) {
            v8f z = {0.f,0.f,0.f,0.f,0.f,0.f,0.f,0.f};
            acc[i][j] = z;
        }

    const int arow = tid >> 1, aseg = (tid & 1) * 16;
    const int brow = tid >> 3, bseg = (tid & 7) * 16;

    auto compute = [&](const unsigned short* sA, const unsigned short* sB) {
        Frag16 af[4], bf[2];
#pragma unroll
        for (int i = 0; i < 4; i++) {
            const unsigned short* p = sA + (wM + i * 16 + l16) * 32 + lh * 8;
            af[i].q[0] = *(const v4u*)(p);
            af[i].q[1] = *(const v4u*)(p + 16);
        }
#pragma unroll
        for (int j = 0; j < 2; j++) {
            const unsigned short* p = sB + wN + j * 16 + l16;
#pragma unroll
            for (int v = 0; v < 8; v++) {
                int k = lh * 8 + (v & 3) * 2 + (v >> 2) * 16;
                bf[j].u[v] = (unsigned int)p[k * 128] | ((unsigned int)p[(k + 1) * 128] << 16);
            }
        }
#pragma unroll
        for (int i = 0; i < 4; i++)
#pragma unroll
            for (int j = 0; j < 2; j++)
                acc[i][j] = wmma_bf16(af[i].v, bf[j].v, acc[i][j]);
    };

#if HAVE_ASYNC_LDS
    // ---- double-buffered async DMA pipeline ----
    auto stage = [&](int kk, int p) {
        const unsigned short* ga = A + (size_t)(bM + arow) * K + kk + aseg;
        unsigned short* sa = lA[p] + arow * 32 + aseg;
        async_b128(ga,     sa);
        async_b128(ga + 8, sa + 8);
        const unsigned short* gb = Bt + (size_t)(kk + brow) * N + bN + bseg;
        unsigned short* sb = lB[p] + brow * 128 + bseg;
        async_b128(gb,     sb);
        async_b128(gb + 8, sb + 8);
    };
    stage(0, 0);
    int p = 0;
    for (int kk = 0; kk < K; kk += 32, p ^= 1) {
        wait_async0();          // this wave's DMA into buffer p done
        __syncthreads();        // everyone's DMA done; prior reads of p^1 done
        if (kk + 32 < K) stage(kk + 32, p ^ 1);   // overlap next DMA w/ WMMA
        compute(lA[p], lB[p]);
    }
#else
    for (int kk = 0; kk < K; kk += 32) {
        {
            const v4u* ga = (const v4u*)(A + (size_t)(bM + arow) * K + kk + aseg);
            v4u* sa = (v4u*)(lA[0] + arow * 32 + aseg);
            sa[0] = ga[0]; sa[1] = ga[1];
            const v4u* gb = (const v4u*)(Bt + (size_t)(kk + brow) * N + bN + bseg);
            v4u* sb = (v4u*)(lB[0] + brow * 128 + bseg);
            sb[0] = gb[0]; sb[1] = gb[1];
            if (kk + 32 < K) {
                __builtin_prefetch(A + (size_t)(bM + arow) * K + kk + 32 + aseg, 0, 1);
                __builtin_prefetch(Bt + (size_t)(kk + 32 + brow) * N + bN + bseg, 0, 1);
            }
        }
        __syncthreads();
        compute(lA[0], lB[0]);
        __syncthreads();
    }
#endif

    // ---- epilogue: C layout lane=col+16*(row/8), vgpr=row%8 ----
#pragma unroll
    for (int j = 0; j < 2; j++) {
        int col = bN + wN + j * 16 + l16;
        float bia = bias[col];
#pragma unroll
        for (int i = 0; i < 4; i++) {
#pragma unroll
            for (int r = 0; r < 8; r++) {
                int row = bM + wM + i * 16 + r + 8 * lh;
                size_t idx = (size_t)row * N + col;
                float v = acc[i][j][r] + bia;
                if (RELU)  v = fmaxf(v, 0.f);
                if (RESID) v += resid[idx];
                if (OUT_BF16) ((unsigned short*)Cout)[idx] = f2bf(v);
                else          ((float*)Cout)[idx] = v;
            }
        }
    }
}

// ---------------------------------------------------------------------------
// RoPE + split: qkv [M,1536] bf16 -> Q,K (roped), V in [B*H, S, 64] bf16
// ---------------------------------------------------------------------------
__global__ __launch_bounds__(256) void rope_split_kernel(
    const unsigned short* __restrict__ qkv,
    unsigned short* __restrict__ Qb,
    unsigned short* __restrict__ Kb,
    unsigned short* __restrict__ Vb)
{
    int idx = blockIdx.x * 256 + threadIdx.x;   // M * 8 * 32
    int i = idx & 31;
    int h = (idx >> 5) & 7;
    int m = idx >> 8;                           // token 0..8191
    int s = m & (SEQ - 1);
    int bb = m >> 12;

    const unsigned short* row = qkv + (size_t)m * (3 * D_MODEL);
    float q0 = bf2f(row[h * 64 + 2 * i]);
    float q1 = bf2f(row[h * 64 + 2 * i + 1]);
    float k0 = bf2f(row[512 + h * 64 + 2 * i]);
    float k1 = bf2f(row[512 + h * 64 + 2 * i + 1]);

    // inv_freq = 10000^(-i/32) = exp(-i * ln(10000)/32)
    float ang = (float)s * __expf(-(float)i * 0.2878231366f);
    float sn, cs;
    __sincosf(ang, &sn, &cs);

    size_t base = (((size_t)(bb * N_HEADS + h)) * SEQ + s) * HEAD_DIM;
    Qb[base + 2 * i]     = f2bf(q0 * cs - q1 * sn);
    Qb[base + 2 * i + 1] = f2bf(q0 * sn + q1 * cs);
    Kb[base + 2 * i]     = f2bf(k0 * cs - k1 * sn);
    Kb[base + 2 * i + 1] = f2bf(k0 * sn + k1 * cs);
    Vb[base + 2 * i]     = row[1024 + h * 64 + 2 * i];
    Vb[base + 2 * i + 1] = row[1024 + h * 64 + 2 * i + 1];
}

// ---------------------------------------------------------------------------
// Flash attention (causal): one wave per 16-query tile of one (b,h).
// S^T = K * Q^T via WMMA; online softmax; O += P*V via WMMA.
// Only the single diagonal block is masked (MASK template).
// ---------------------------------------------------------------------------
template<bool MASK>
__device__ __forceinline__ void flash_block(
    int c, int q0, int lane, int lh, int l16,
    const unsigned short* __restrict__ Kp,
    const unsigned short* __restrict__ Vp,
    const Frag16 (&bQ)[2], unsigned short* __restrict__ ldsPw,
    v8f (&O)[4], float& m_i, float& l_i)
{
    // ---- S^T tiles (rows = kv, cols = q); Q already scaled by 1/sqrt(64) ----
    v8f cS[2];
#pragma unroll
    for (int tt = 0; tt < 2; tt++) {
        Frag16 aK[2];
#pragma unroll
        for (int t = 0; t < 2; t++) {
            const unsigned short* p =
                Kp + (size_t)(c + tt * 16 + l16) * HEAD_DIM + t * 32 + lh * 8;
            aK[t].q[0] = *(const v4u*)(p);
            aK[t].q[1] = *(const v4u*)(p + 16);
        }
        v8f z = {0.f,0.f,0.f,0.f,0.f,0.f,0.f,0.f};
        v8f s0 = wmma_bf16(aK[0].v, bQ[0].v, z);
        cS[tt]  = wmma_bf16(aK[1].v, bQ[1].v, s0);
    }

    int qg = q0 + l16;
    float mx = -3.0e38f;
#pragma unroll
    for (int tt = 0; tt < 2; tt++) {
#pragma unroll
        for (int r = 0; r < 8; r++) {
            float sv = cS[tt][r];
            if (MASK) {
                int kvi = c + tt * 16 + r + 8 * lh;
                sv = (kvi <= qg) ? sv : -3.0e38f;
                cS[tt][r] = sv;
            }
            mx = fmaxf(mx, sv);
        }
    }
    mx = fmaxf(mx, __shfl_xor(mx, 16));
    float m_new = fmaxf(m_i, mx);
    float alpha = __expf(m_i - m_new);

    // ---- p = exp(s - m_new); stash to LDS in A layout [q][kv] ----
    float ls = 0.f;
#pragma unroll
    for (int tt = 0; tt < 2; tt++) {
#pragma unroll
        for (int r = 0; r < 8; r++) {
            float p = __expf(cS[tt][r] - m_new);
            ls += p;
            ldsPw[l16 * 32 + tt * 16 + r + 8 * lh] = f2bf(p);
        }
    }
    ls += __shfl_xor(ls, 16);
    l_i = l_i * alpha + ls;
    m_i = m_new;

    // ---- rescale O (broadcast alpha per output row) ----
    float af[8];
#pragma unroll
    for (int r = 0; r < 8; r++) af[r] = __shfl(alpha, r + 8 * lh);
#pragma unroll
    for (int g = 0; g < 4; g++)
#pragma unroll
        for (int r = 0; r < 8; r++) O[g][r] *= af[r];

    // ---- P (16x32) as A fragment from LDS ----
    Frag16 aP;
    const unsigned short* pp = ldsPw + l16 * 32 + lh * 8;
    aP.q[0] = *(const v4u*)(pp);
    aP.q[1] = *(const v4u*)(pp + 16);

    // ---- O += P * V : contiguous V rows as B fragments ----
#pragma unroll
    for (int g = 0; g < 4; g++) {
        Frag16 bV;
        const unsigned short* vp = Vp + (size_t)c * HEAD_DIM + g * 16 + l16;
#pragma unroll
        for (int v = 0; v < 8; v++) {
            int k = lh * 8 + (v & 3) * 2 + (v >> 2) * 16;
            bV.u[v] = (unsigned int)vp[(size_t)k * HEAD_DIM] | ((unsigned int)vp[(size_t)(k + 1) * HEAD_DIM] << 16);
        }
        O[g] = wmma_bf16(aP.v, bV.v, O[g]);
    }
}

__global__ __launch_bounds__(256) void flash_kernel(
    const unsigned short* __restrict__ Qb,
    const unsigned short* __restrict__ Kb,
    const unsigned short* __restrict__ Vb,
    unsigned short* __restrict__ ctx)  // [B*S, 512], col = h*64 + d
{
    __shared__ __align__(16) unsigned short ldsP[8][16 * 32];

    int wave = threadIdx.x >> 5;
    int lane = threadIdx.x & 31;
    int lh   = lane >> 4;
    int l16  = lane & 15;

    int tile = blockIdx.x * 8 + wave;     // 0 .. 4095
    int qt = tile & 255;
    int bh = tile >> 8;                   // 0..15
    int q0 = qt * 16;

    const unsigned short* Qp = Qb + (size_t)bh * SEQ * HEAD_DIM;
    const unsigned short* Kp = Kb + (size_t)bh * SEQ * HEAD_DIM;
    const unsigned short* Vp = Vb + (size_t)bh * SEQ * HEAD_DIM;

    // Q^T as B fragments, pre-scaled by 1/sqrt(HEAD_DIM)
    Frag16 bQ[2];
#pragma unroll
    for (int t = 0; t < 2; t++) {
#pragma unroll
        for (int v = 0; v < 8; v++) {
            int k = t * 32 + lh * 8 + (v & 3) * 2 + (v >> 2) * 16;
            unsigned int lo = f2bf(0.125f * bf2f(Qp[(size_t)(q0 + l16) * HEAD_DIM + k]));
            unsigned int hi = f2bf(0.125f * bf2f(Qp[(size_t)(q0 + l16) * HEAD_DIM + k + 1]));
            bQ[t].u[v] = lo | (hi << 16);
        }
    }

    v8f O[4];
#pragma unroll
    for (int g = 0; g < 4; g++) {
        v8f z = {0.f,0.f,0.f,0.f,0.f,0.f,0.f,0.f};
        O[g] = z;
    }
    float m_i = -3.0e38f, l_i = 0.f;

    // full (unmasked) blocks, then exactly one diagonal (masked) block
    int nfull = q0 >> 5;
    for (int j = 0; j < nfull; j++)
        flash_block<false>(j * 32, q0, lane, lh, l16, Kp, Vp, bQ, ldsP[wave], O, m_i, l_i);
    flash_block<true>(nfull * 32, q0, lane, lh, l16, Kp, Vp, bQ, ldsP[wave], O, m_i, l_i);

    // ---- normalize and write ctx (bf16) ----
    float linv = 1.f / l_i;
    float lf[8];
#pragma unroll
    for (int r = 0; r < 8; r++) lf[r] = __shfl(linv, r + 8 * lh);

    int b = bh >> 3, h = bh & 7;
#pragma unroll
    for (int g = 0; g < 4; g++) {
#pragma unroll
        for (int r = 0; r < 8; r++) {
            int qrow = q0 + r + 8 * lh;
            size_t idx = ((size_t)(b * SEQ + qrow)) * D_MODEL + h * 64 + g * 16 + l16;
            ctx[idx] = f2bf(O[g][r] * lf[r]);
        }
    }
}

// ---------------------------------------------------------------------------
// Host driver
// ---------------------------------------------------------------------------
extern "C" void kernel_launch(void* const* d_in, const int* in_sizes, int n_in,
                              void* d_out, int out_size, void* d_ws, size_t ws_size,
                              hipStream_t stream) {
    const float* src      = (const float*)d_in[0];
    const float* ln_w     = (const float*)d_in[1];
    const float* ln_b     = (const float*)d_in[2];
    const float* wqkv_w   = (const float*)d_in[3];
    const float* wqkv_b   = (const float*)d_in[4];
    const float* out_w    = (const float*)d_in[5];
    const float* out_b    = (const float*)d_in[6];
    const float* ffn_ln_w = (const float*)d_in[7];
    const float* ffn_ln_b = (const float*)d_in[8];
    const float* ff1_w    = (const float*)d_in[9];
    const float* ff1_b    = (const float*)d_in[10];
    const float* ff2_w    = (const float*)d_in[11];
    const float* ff2_b    = (const float*)d_in[12];

    char* base = (char*)d_ws;
    size_t off = 0;
    auto alloc = [&](size_t bytes) -> void* {
        void* p = base + off;
        off = (off + bytes + 255) & ~(size_t)255;
        return p;
    };

    unsigned short* WtQKV = (unsigned short*)alloc((size_t)512 * 1536 * 2);
    unsigned short* WtOUT = (unsigned short*)alloc((size_t)512 * 512 * 2);
    unsigned short* WtFF1 = (unsigned short*)alloc((size_t)512 * 2048 * 2);
    unsigned short* WtFF2 = (unsigned short*)alloc((size_t)2048 * 512 * 2);
    unsigned short* A1    = (unsigned short*)alloc((size_t)MROWS * 512 * 2);
    unsigned short* QKV   = (unsigned short*)alloc((size_t)MROWS * 1536 * 2);
    unsigned short* Qbuf  = (unsigned short*)alloc((size_t)MROWS * 512 * 2);
    unsigned short* Kbuf  = (unsigned short*)alloc((size_t)MROWS * 512 * 2);
    unsigned short* Vbuf  = (unsigned short*)alloc((size_t)MROWS * 512 * 2);
    unsigned short* CTX   = (unsigned short*)alloc((size_t)MROWS * 512 * 2);
    float*          SRC2  = (float*)         alloc((size_t)MROWS * 512 * 4);
    unsigned short* A2    = (unsigned short*)alloc((size_t)MROWS * 512 * 2);
    unsigned short* H1    = (unsigned short*)alloc((size_t)MROWS * 2048 * 2);

    // 1) weight conversion + transpose to bf16 [K][N]
    cvtT_kernel<<<(1536 * 512 + 255) / 256, 256, 0, stream>>>(wqkv_w, WtQKV, 1536, 512);
    cvtT_kernel<<<(512 * 512 + 255) / 256, 256, 0, stream>>>(out_w, WtOUT, 512, 512);
    cvtT_kernel<<<(2048 * 512 + 255) / 256, 256, 0, stream>>>(ff1_w, WtFF1, 2048, 512);
    cvtT_kernel<<<(512 * 2048 + 255) / 256, 256, 0, stream>>>(ff2_w, WtFF2, 512, 2048);

    // 2) LN1 -> bf16
    ln_kernel<<<MROWS / 8, 256, 0, stream>>>(src, ln_w, ln_b, A1, MROWS);

    // 3) QKV GEMM (M=8192, N=1536, K=512) -> bf16 (+bias)
    gemm_bf16_kernel<false, false, true><<<dim3(1536 / 128, MROWS / 128), 256, 0, stream>>>(
        A1, WtQKV, wqkv_b, nullptr, QKV, MROWS, 1536, 512);

    // 4) RoPE + split into [B*H, S, 64]
    rope_split_kernel<<<(MROWS * N_HEADS * 32) / 256, 256, 0, stream>>>(QKV, Qbuf, Kbuf, Vbuf);

    // 5) causal flash attention -> CTX bf16 [M, 512]
    flash_kernel<<<(BATCH * N_HEADS * (SEQ / 16)) / 8, 256, 0, stream>>>(Qbuf, Kbuf, Vbuf, CTX);

    // 6) out-proj GEMM + residual(src) -> SRC2 fp32
    gemm_bf16_kernel<false, true, false><<<dim3(512 / 128, MROWS / 128), 256, 0, stream>>>(
        CTX, WtOUT, out_b, src, SRC2, MROWS, 512, 512);

    // 7) LN2 -> bf16
    ln_kernel<<<MROWS / 8, 256, 0, stream>>>(SRC2, ffn_ln_w, ffn_ln_b, A2, MROWS);

    // 8) FF1 GEMM + ReLU -> bf16 (M=8192, N=2048, K=512)
    gemm_bf16_kernel<true, false, true><<<dim3(2048 / 128, MROWS / 128), 256, 0, stream>>>(
        A2, WtFF1, ff1_b, nullptr, H1, MROWS, 2048, 512);

    // 9) FF2 GEMM + residual(SRC2) -> d_out fp32 (M=8192, N=512, K=2048)
    gemm_bf16_kernel<false, true, false><<<dim3(512 / 128, MROWS / 128), 256, 0, stream>>>(
        H1, WtFF2, ff2_b, SRC2, (float*)d_out, MROWS, 512, 2048);
}
